// LSTMCell_884763263700
// MI455X (gfx1250) — compile-verified, matrix operand order checked by reading, not run on hardware
//
#include <hip/hip_runtime.h>

typedef __attribute__((ext_vector_type(16))) __bf16 bf16x16;
typedef __attribute__((ext_vector_type(8)))  __bf16 bf16x8;
typedef __attribute__((ext_vector_type(8)))  float  f32x8;

#define B_ROWS 8192
#define DIN 1024
#define DOUT 1024
#define KDIM 2048           // DIN + DOUT, fused K
#define MT 128              // M tile per workgroup
#define JT 32               // j columns per workgroup (x4 gates => N tile 128)
#define KT 32               // K step (matches wmma 16x16x32)
#define LDS_STRIDE 40       // bf16 elems per LDS row (32 + 8 pad, conflict-free b128 reads)
#define STAGE_ELEMS (128 * LDS_STRIDE)

__device__ __forceinline__ unsigned short f2bf(float f) {
    unsigned int u = __builtin_bit_cast(unsigned int, f);
    u += 0x7FFFu + ((u >> 16) & 1u);   // round-to-nearest-even
    return (unsigned short)(u >> 16);
}

__device__ __forceinline__ float sigmoidf_(float x) {
    return 1.0f / (1.0f + __expf(-x));
}

__device__ __forceinline__ bf16x16 ld_frag(const unsigned short* base, int off) {
    bf16x8 lo = *(const bf16x8*)(base + off);
    bf16x8 hi = *(const bf16x8*)(base + off + 16);
    return __builtin_shufflevector(lo, hi, 0, 1, 2, 3, 4, 5, 6, 7,
                                            8, 9, 10, 11, 12, 13, 14, 15);
}

// reuse_a must be an ICE at the builtin call -> template parameter
template <bool RA>
__device__ __forceinline__ f32x8 wmma_bf16(bf16x16 a, bf16x16 b, f32x8 c) {
    return __builtin_amdgcn_wmma_f32_16x16x32_bf16(false, a, false, b,
                                                   (short)0, c, RA, false);
}

// CDNA5 async global->LDS copy (ASYNCcnt tracked, no data VGPRs).
__device__ __forceinline__ void async_b128(unsigned lds_byte, const void* g) {
    asm volatile("global_load_async_to_lds_b128 %0, %1, off"
                 :: "v"(lds_byte), "v"(g)
                 : "memory");
}

__device__ __forceinline__ void wait_async0() {
    asm volatile("s_wait_asynccnt 0x0" ::: "memory");
}

// ---------------- packing kernels ----------------

// A[row][k] bf16: k<1024 from x, k>=1024 from h
__global__ void pack_a_kernel(const float* __restrict__ x,
                              const float* __restrict__ h,
                              unsigned short* __restrict__ A) {
    const int idx = (blockIdx.x * 256 + threadIdx.x) * 4;
    const int row = idx >> 11;          // /2048
    const int col = idx & 2047;
    const float* src = (col < DIN) ? (x + (size_t)row * DIN + col)
                                   : (h + (size_t)row * DOUT + (col - DIN));
    float4 v = *(const float4*)src;
    union { unsigned short s[4]; uint2 u; } o;
    o.s[0] = f2bf(v.x); o.s[1] = f2bf(v.y); o.s[2] = f2bf(v.z); o.s[3] = f2bf(v.w);
    *(uint2*)(A + idx) = o.u;
}

struct WPtrs { const float* p[8]; };  // Wxi,Whi,Wxf,Whf,Wxg,Whg,Wxo,Who

// Wt[n][k] bf16 with n = gate*1024 + j; k slot 0..1023 = Wx rows, 1024..2047 = Wh rows.
// 32x32 LDS tile transpose: coalesced reads along j, coalesced writes along k.
__global__ void pack_w_kernel(WPtrs P, unsigned short* __restrict__ Wt) {
    __shared__ float tile[32][33];
    const int z = blockIdx.z;
    const int gate = z >> 1;
    const int isH = z & 1;
    const float* __restrict__ src = P.p[z];
    const int j0t = blockIdx.x * 32;
    const int k0t = blockIdx.y * 32;
#pragma unroll
    for (int r = 0; r < 4; ++r) {
        const int ty = threadIdx.y + r * 8;
        tile[ty][threadIdx.x] = src[(size_t)(k0t + ty) * DOUT + j0t + threadIdx.x];
    }
    __syncthreads();
#pragma unroll
    for (int r = 0; r < 4; ++r) {
        const int ty = threadIdx.y + r * 8;
        const int n = gate * DOUT + j0t + ty;
        const int ks = isH * DIN + k0t + threadIdx.x;
        Wt[(size_t)n * KDIM + ks] = f2bf(tile[threadIdx.x][ty]);
    }
}

struct BPtrs { const float* bx[4]; const float* bh[4]; };

__global__ void pack_bias_kernel(BPtrs P, float* __restrict__ bias) {
    const int j = blockIdx.x * 256 + threadIdx.x;   // 0..4095
    const int g = j >> 10, jj = j & 1023;
    bias[j] = P.bx[g][jj] + P.bh[g][jj];
}

// ---------------- fused GEMM + LSTM gate kernel ----------------

__global__ __launch_bounds__(256) void lstm_gemm_kernel(
    const unsigned short* __restrict__ A,    // [8192][2048] bf16
    const unsigned short* __restrict__ Wt,   // [4096][2048] bf16 (n-major)
    const float* __restrict__ bias,          // [4096]
    const float* __restrict__ cin,           // [8192][1024]
    float* __restrict__ hout,                // [8192][1024]
    float* __restrict__ cout_) {             // [8192][1024]
    __shared__ __align__(16) unsigned short smem[4 * STAGE_ELEMS]; // A[2], B[2] stages

    const int m0 = blockIdx.x * MT;
    const int j0 = blockIdx.y * JT;
    const int t = threadIdx.x;
    const int lane = t & 31;
    const int wv = t >> 5;
    const int laneLo = lane & 15;
    const int laneHi = lane >> 4;

    // global -> LDS staging map: thread t covers (row = t/4 [+64], kk = (t%4)*8)
    const int lrow = t >> 2;
    const int lkk = (t & 3) * 8;
    const int r1 = lrow + 64;
    const unsigned short* gA0 = A + (size_t)(m0 + lrow) * KDIM + lkk;
    const unsigned short* gA1 = A + (size_t)(m0 + r1) * KDIM + lkk;
    const int n0 = (lrow >> 5) * DOUT + j0 + (lrow & 31);
    const int n1 = (r1 >> 5) * DOUT + j0 + (r1 & 31);
    const unsigned short* gB0 = Wt + (size_t)n0 * KDIM + lkk;
    const unsigned short* gB1 = Wt + (size_t)n1 * KDIM + lkk;
    const int s0 = lrow * LDS_STRIDE + lkk;
    const int s1 = r1 * LDS_STRIDE + lkk;

    // LDS byte addresses for async-to-LDS staging (stage 0 homes)
    const unsigned sbase = (unsigned)(uintptr_t)(&smem[0]);
    const unsigned aL0 = sbase + (unsigned)(s0 * 2);
    const unsigned aL1 = sbase + (unsigned)(s1 * 2);
    const unsigned bL0 = sbase + (unsigned)((2 * STAGE_ELEMS + s0) * 2);
    const unsigned bL1 = sbase + (unsigned)((2 * STAGE_ELEMS + s1) * 2);

    f32x8 acc[4][2] = {};
    const int aOff = (wv * 16 + laneLo) * LDS_STRIDE + laneHi * 8;

    // one k-step of 8 WMMAs; A fragment shared across all 8 -> reuse-A hint on 2..8
    auto compute_step = [&](const unsigned short* As, const unsigned short* Bs) {
        bf16x16 aF = ld_frag(As, aOff);
#define GATE_STEP(G, S, RA)                                                     \
        do {                                                                    \
            bf16x16 bF = ld_frag(                                              \
                Bs, ((G) * 32 + (S) * 16 + laneLo) * LDS_STRIDE + laneHi * 8); \
            acc[G][S] = wmma_bf16<RA>(aF, bF, acc[G][S]);                       \
        } while (0)
        GATE_STEP(0, 0, false);
        GATE_STEP(0, 1, true);
        GATE_STEP(1, 0, true);
        GATE_STEP(1, 1, true);
        GATE_STEP(2, 0, true);
        GATE_STEP(2, 1, true);
        GATE_STEP(3, 0, true);
        GATE_STEP(3, 1, true);
#undef GATE_STEP
    };

    // prologue: async-stage k-step 0 into buffer 0
    async_b128(aL0, gA0);
    async_b128(aL1, gA1);
    async_b128(bL0, gB0);
    async_b128(bL1, gB1);
    wait_async0();
    __syncthreads();

    const int NK = KDIM / KT;   // 64
    unsigned curOff = 0;        // elem offset of current stage (0 or STAGE_ELEMS)

#pragma unroll 1
    for (int kt = 0; kt < NK - 1; ++kt) {
        const int koff = (kt + 1) * KT;
        const unsigned nxtOff = curOff ^ (unsigned)STAGE_ELEMS;
        const unsigned nb = nxtOff * 2;   // byte offset of next stage
        // stream next k-step into the other buffer while we compute
        async_b128(aL0 + nb, gA0 + koff);
        async_b128(aL1 + nb, gA1 + koff);
        async_b128(bL0 + nb, gB0 + koff);
        async_b128(bL1 + nb, gB1 + koff);

        compute_step(&smem[curOff], &smem[2 * STAGE_ELEMS + curOff]);

        // own async writes must land before the barrier publishes the buffer
        wait_async0();
        __syncthreads();
        curOff = nxtOff;
    }
    // peeled final k-step (no prefetch, no barrier needed afterwards)
    compute_step(&smem[curOff], &smem[2 * STAGE_ELEMS + curOff]);

    // fused LSTM epilogue: C/D layout -> element (m = v + 8*laneHi, n = laneLo)
    const int mBase = m0 + wv * 16 + laneHi * 8;
#pragma unroll
    for (int s = 0; s < 2; ++s) {
        const int j = j0 + s * 16 + laneLo;
        const float bi = bias[j];
        const float bf = bias[DOUT + j];
        const float bg = bias[2 * DOUT + j];
        const float bo = bias[3 * DOUT + j];
#pragma unroll
        for (int v = 0; v < 8; ++v) {
            const int m = mBase + v;
            const size_t o = (size_t)m * DOUT + j;
            const float ig = sigmoidf_(acc[0][s][v] + bi);
            const float fg = sigmoidf_(acc[1][s][v] + bf);
            const float gg = tanhf(acc[2][s][v] + bg);
            const float og = sigmoidf_(acc[3][s][v] + bo);
            const float cn = fg * cin[o] + ig * gg;
            hout[o] = og * tanhf(cn);
            cout_[o] = cn;
        }
    }
}

// ---------------- launch ----------------

extern "C" void kernel_launch(void* const* d_in, const int* in_sizes, int n_in,
                              void* d_out, int out_size, void* d_ws, size_t ws_size,
                              hipStream_t stream) {
    const float* x = (const float*)d_in[0];
    const float* h = (const float*)d_in[1];
    const float* c = (const float*)d_in[2];

    WPtrs wp;
    wp.p[0] = (const float*)d_in[3];   // Wxi
    wp.p[1] = (const float*)d_in[4];   // Whi
    wp.p[2] = (const float*)d_in[7];   // Wxf
    wp.p[3] = (const float*)d_in[8];   // Whf
    wp.p[4] = (const float*)d_in[11];  // Wxg
    wp.p[5] = (const float*)d_in[12];  // Whg
    wp.p[6] = (const float*)d_in[15];  // Wxo
    wp.p[7] = (const float*)d_in[16];  // Who

    BPtrs bp;
    bp.bx[0] = (const float*)d_in[5];  bp.bh[0] = (const float*)d_in[6];
    bp.bx[1] = (const float*)d_in[9];  bp.bh[1] = (const float*)d_in[10];
    bp.bx[2] = (const float*)d_in[13]; bp.bh[2] = (const float*)d_in[14];
    bp.bx[3] = (const float*)d_in[17]; bp.bh[3] = (const float*)d_in[18];

    const size_t aBytes = (size_t)B_ROWS * KDIM * sizeof(unsigned short);   // 33.5 MB
    const size_t wBytes = (size_t)4 * DOUT * KDIM * sizeof(unsigned short); // 16.8 MB
    unsigned short* Abf = (unsigned short*)d_ws;
    unsigned short* Wt = (unsigned short*)((char*)d_ws + aBytes);
    float* bias = (float*)((char*)d_ws + aBytes + wBytes);

    float* houtp = (float*)d_out;
    float* coutp = houtp + (size_t)B_ROWS * DOUT;

    pack_a_kernel<<<(B_ROWS * KDIM / 4) / 256, 256, 0, stream>>>(x, h, Abf);
    pack_w_kernel<<<dim3(32, 32, 8), dim3(32, 8), 0, stream>>>(wp, Wt);
    pack_bias_kernel<<<16, 256, 0, stream>>>(bp, bias);
    lstm_gemm_kernel<<<dim3(B_ROWS / MT, DOUT / JT), 256, 0, stream>>>(
        Abf, Wt, bias, c, houtp, coutp);
}
